// AxisMoE_62766652064416
// MI455X (gfx1250) — matrix operand order repaired and checked by reading, not compiled, not run on hardware
//
#include <hip/hip_runtime.h>
#include <hip/hip_bf16.h>

// ---- problem dims (fixed by the reference) ----
constexpr int BDIM = 8;
constexpr int SDIM = 4096;
constexpr int DDIM = 1024;
constexpr int DADIM = 128;
constexpr int EDIM = 8;
constexpr int TOK = BDIM * SDIM;          // 32768 tokens

// ---- GEMM tiling ----
constexpr int BM = 128;                   // gathered-token rows per block
constexpr int BN = 64;                    // output features per block
constexpr int BK = 32;                    // K step == WMMA K
constexpr int LDP = BK + 8;               // LDS row stride in halves (80B: keeps 16B alignment)

typedef __attribute__((ext_vector_type(16))) __bf16 v16bf;
typedef __attribute__((ext_vector_type(8)))  __bf16 v8bf;
typedef __attribute__((ext_vector_type(4)))  __bf16 v4bf;
typedef __attribute__((ext_vector_type(8)))  float  v8f;
typedef __attribute__((ext_vector_type(4)))  float  f4;

struct u2 { unsigned int x, y; };

// f32x4 -> bf16x4 as one 8-byte payload; element-wise convertvector lets the
// backend fuse pairs into v_cvt_pk_bf16_f32 (the manual shift/or idiom did not).
__device__ __forceinline__ u2 pk4(f4 v) {
  v4bf c = __builtin_convertvector(v, v4bf);
  return __builtin_bit_cast(u2, c);
}

// =====================================================================
// Stage 1: gating — logits, softmax, top-2, routing lists, entropy loss
// One wave (32 lanes) per token; 8 waves per 256-thread block.
// =====================================================================
__global__ void axis_moe_gate(const float* __restrict__ h,
                              const float* __restrict__ a,
                              const float* __restrict__ gW,
                              const float* __restrict__ gb,
                              int*   __restrict__ cnt,     // [E]
                              int*   __restrict__ idxl,    // [E][TOK]
                              float* __restrict__ wgtl,    // [E][TOK]
                              float* __restrict__ ent_acc) // &d_out[B*S*D]
{
  const int wave = threadIdx.x >> 5;
  const int lane = threadIdx.x & 31;
  const int token = blockIdx.x * 8 + wave;
  const int batch = token >> 12;                       // token / 4096

  const float* hrow = h + (size_t)token * DDIM;
  const float* arow = a + batch * DADIM;

  float logit[EDIM];
#pragma unroll
  for (int e = 0; e < EDIM; ++e) {
    const float* w = gW + e * (DDIM + DADIM);
    float acc = 0.f;
#pragma unroll 2
    for (int k = lane * 4; k < DDIM; k += 128) {       // float4 per lane
      f4 x = *(const f4*)(hrow + k);
      f4 g = *(const f4*)(w + k);
      acc += x[0]*g[0] + x[1]*g[1] + x[2]*g[2] + x[3]*g[3];
    }
    {
      f4 x = *(const f4*)(arow + lane * 4);            // DA=128 = 32 lanes * 4
      f4 g = *(const f4*)(w + DDIM + lane * 4);
      acc += x[0]*g[0] + x[1]*g[1] + x[2]*g[2] + x[3]*g[3];
    }
#pragma unroll
    for (int off = 16; off > 0; off >>= 1) acc += __shfl_xor(acc, off, 32);
    logit[e] = acc + gb[e];
  }

  float mx = logit[0];
#pragma unroll
  for (int e = 1; e < EDIM; ++e) mx = fmaxf(mx, logit[e]);
  float g[EDIM], s = 0.f;
#pragma unroll
  for (int e = 0; e < EDIM; ++e) { g[e] = __expf(logit[e] - mx); s += g[e]; }
  float inv = 1.f / s;
#pragma unroll
  for (int e = 0; e < EDIM; ++e) g[e] *= inv;

  float ent = 0.f;
#pragma unroll
  for (int e = 0; e < EDIM; ++e) ent += g[e] * __logf(g[e] + 1e-10f);

  int i1 = 0;
#pragma unroll
  for (int e = 1; e < EDIM; ++e) if (g[e] > g[i1]) i1 = e;
  int i2 = (i1 == 0) ? 1 : 0;
#pragma unroll
  for (int e = 0; e < EDIM; ++e) if (e != i1 && g[e] > g[i2]) i2 = e;
  const float wsum = g[i1] + g[i2];
  const float w1 = g[i1] / wsum, w2 = g[i2] / wsum;

  if (lane == 0) {
    atomicAdd(ent_acc, -ent * (1.0f / (float)TOK));
    int p1 = atomicAdd(&cnt[i1], 1);
    idxl[i1 * TOK + p1] = token;  wgtl[i1 * TOK + p1] = w1;
    int p2 = atomicAdd(&cnt[i2], 1);
    idxl[i2 * TOK + p2] = token;  wgtl[i2 * TOK + p2] = w2;
  }
}

// =====================================================================
// Stage 2: per-expert gathered GEMM, bf16 WMMA, ping-pong LDS pipeline.
// grid = (DDIM/BN, TOK/BM, EDIM); 256 threads = 8 waves in a 4x2 grid;
// each wave computes a 32x32 tile: 2 A frags x 2 B frags x 4 WMMAs/K-step.
// =====================================================================
__global__ void __launch_bounds__(256)
axis_moe_expert_gemm(const float* __restrict__ h,
                     const float* __restrict__ expW,   // [E][D][D] (out,in)
                     const float* __restrict__ expB,   // [E][D]
                     const int*   __restrict__ cnt,    // [E]
                     const int*   __restrict__ idxl,   // [E][TOK]
                     const float* __restrict__ wgtl,   // [E][TOK]
                     float*       __restrict__ out)    // [TOK][D], pre-zeroed
{
  __shared__ __align__(16) unsigned short As[2][BM][LDP];
  __shared__ __align__(16) unsigned short Bs[2][BN][LDP];
  __shared__ int rowidx[BM];

  const int e  = blockIdx.z;
  const int m0 = blockIdx.y * BM;
  const int n0 = blockIdx.x * BN;
  const int cntE = cnt[e];
  if (m0 >= cntE) return;                       // block-uniform early exit

  const int tid  = threadIdx.x;
  const int wave = tid >> 5;
  const int lane = tid & 31;
  const int lmod = lane & 15;
  const int lhi  = lane >> 4;                   // 0: lanes 0-15, 1: lanes 16-31
  const int wr   = wave >> 1;                   // 0..3 -> M strip of 32 rows
  const int wc   = wave & 1;                    // 0..1 -> N strip of 32 cols

  const float* We   = expW + (size_t)e * DDIM * DDIM;
  const int*   midx = idxl + e * TOK;

  // ---- hoist the gather: row indices loaded ONCE (clamped; extra rows are
  //      don't-care since the epilogue skips p >= cntE) ----
  if (tid < BM) {
    int p = m0 + tid;
    rowidx[tid] = midx[p < cntE ? p : (cntE - 1)];
  }
  __syncthreads();

  // fixed per-thread staging slots:
  // A: 128x32 f32 = 1024 float4 -> 4 per thread; B: 64x32 = 512 float4 -> 2.
  int ar[4], ak[4], abase[4];
#pragma unroll
  for (int j = 0; j < 4; ++j) {
    const int i = tid + j * 256;
    ar[j] = i >> 3;                             // 8 float4 per row
    ak[j] = (i & 7) << 2;
    abase[j] = rowidx[ar[j]];
  }
  int bn_[2], bk_[2];
#pragma unroll
  for (int j = 0; j < 2; ++j) {
    const int i = tid + j * 256;
    bn_[j] = i >> 3;
    bk_[j] = (i & 7) << 2;
  }

  f4 areg[4], breg[2];
  auto fetch = [&](int kk) {
#pragma unroll
    for (int j = 0; j < 4; ++j)
      areg[j] = *(const f4*)(h + (size_t)abase[j] * DDIM + kk + ak[j]);
#pragma unroll
    for (int j = 0; j < 2; ++j)
      breg[j] = *(const f4*)(We + (size_t)(n0 + bn_[j]) * DDIM + kk + bk_[j]);
  };
  auto stash = [&](int buf) {
#pragma unroll
    for (int j = 0; j < 4; ++j)
      *(u2*)&As[buf][ar[j]][ak[j]] = pk4(areg[j]);   // 2x v_cvt_pk_bf16_f32 + ds_store_b64
#pragma unroll
    for (int j = 0; j < 2; ++j)
      *(u2*)&Bs[buf][bn_[j]][bk_[j]] = pk4(breg[j]);
  };

  v8f c[2][2];
#pragma unroll
  for (int mi = 0; mi < 2; ++mi)
#pragma unroll
    for (int ni = 0; ni < 2; ++ni)
#pragma unroll
      for (int v = 0; v < 8; ++v) c[mi][ni][v] = 0.f;

  fetch(0);
  stash(0);
  __syncthreads();

  constexpr int KIT = DDIM / BK;                // 32
  for (int it = 0; it < KIT; ++it) {
    const int cur = it & 1, nxt = cur ^ 1;
    const bool more = (it + 1 < KIT);
    if (more) {
      fetch((it + 1) * BK);                     // global b128 loads fly under WMMAs
      __builtin_prefetch(We + (size_t)(n0 + (tid & 63)) * DDIM + (it + 2) * BK, 0, 1);
    }

    // A fragments (ISA: lanes<16 K0-7 & 16-23; lanes>=16 K8-15 & 24-31)
    const int ko = lhi ? 8 : 0;
    v16bf af[2];
#pragma unroll
    for (int mi = 0; mi < 2; ++mi) {
      const int m = (wr << 5) + (mi << 4) + lmod;
      v8bf alo = *(const v8bf*)&As[cur][m][ko];
      v8bf ahi = *(const v8bf*)&As[cur][m][ko + 16];
      af[mi] = __builtin_shufflevector(alo, ahi, 0,1,2,3,4,5,6,7,8,9,10,11,12,13,14,15);
    }
    // B fragments (lane col = lmod; lanes<16 K0-7 & 16-23; lanes>=16 K8-15 & 24-31)
    const int kb = lhi ? 8 : 0;
    v16bf bf[2];
#pragma unroll
    for (int ni = 0; ni < 2; ++ni) {
      const int n = (wc << 5) + (ni << 4) + lmod;
      v8bf blo = *(const v8bf*)&Bs[cur][n][kb];
      v8bf bhi = *(const v8bf*)&Bs[cur][n][kb + 16];
      bf[ni] = __builtin_shufflevector(blo, bhi, 0,1,2,3,4,5,6,7,8,9,10,11,12,13,14,15);
    }
#pragma unroll
    for (int mi = 0; mi < 2; ++mi)
#pragma unroll
      for (int ni = 0; ni < 2; ++ni)
        c[mi][ni] = __builtin_amdgcn_wmma_f32_16x16x32_bf16(
            false, af[mi], false, bf[ni], (short)0, c[mi][ni], false, false);

    if (more) stash(nxt);                       // writes other buffer: no race
    __syncthreads();
  }

  // ---- epilogue: (acc + bias) * gate_weight, scatter-add to out ----
  const float* be = expB + e * DDIM;
#pragma unroll
  for (int mi = 0; mi < 2; ++mi) {
#pragma unroll
    for (int ni = 0; ni < 2; ++ni) {
      const int n = n0 + (wc << 5) + (ni << 4) + lmod;
      const float bias = be[n];
#pragma unroll
      for (int v = 0; v < 8; ++v) {
        const int mf = v + (lhi ? 8 : 0);       // C layout: M = vgpr + 8*(lane>=16)
        const int p  = m0 + (wr << 5) + (mi << 4) + mf;
        if (p < cntE) {
          const int trow = midx[p];
          const float gw = wgtl[e * TOK + p];
          atomicAdd(&out[(size_t)trow * DDIM + n], (c[mi][ni][v] + bias) * gw);
        }
      }
    }
  }
}

// =====================================================================
extern "C" void kernel_launch(void* const* d_in, const int* in_sizes, int n_in,
                              void* d_out, int out_size, void* d_ws, size_t ws_size,
                              hipStream_t stream) {
  const float* h   = (const float*)d_in[0];
  const float* a   = (const float*)d_in[1];
  const float* gW  = (const float*)d_in[2];
  const float* gb  = (const float*)d_in[3];
  const float* eW  = (const float*)d_in[4];
  const float* eb  = (const float*)d_in[5];
  float* out = (float*)d_out;

  // workspace: [0,64) counters, then idx lists, then weight lists (~2 MB)
  int*   cnt  = (int*)d_ws;
  int*   idxl = (int*)((char*)d_ws + 64);
  float* wgtl = (float*)((char*)d_ws + 64 + sizeof(int) * (size_t)EDIM * TOK);

  hipMemsetAsync(d_out, 0, sizeof(float) * (size_t)out_size, stream);
  hipMemsetAsync(d_ws, 0, 64, stream);

  float* ent_acc = out + (size_t)BDIM * SDIM * DDIM;   // out[..], entropy, stability

  axis_moe_gate<<<TOK / 8, 256, 0, stream>>>(h, a, gW, gb, cnt, idxl, wgtl, ent_acc);

  dim3 grid(DDIM / BN, TOK / BM, EDIM);                // (16, 256, 8)
  axis_moe_expert_gemm<<<grid, 256, 0, stream>>>(h, eW, eb, cnt, idxl, wgtl, out);
}
